// MoeBlock_1039382085731
// MI455X (gfx1250) — matrix operand order repaired and checked by reading, not compiled, not run on hardware
//
#include <hip/hip_runtime.h>
#include <hip/hip_bf16.h>

// ---------------- problem constants ----------------
#define G  10
#define S  2048
#define MD 2560
#define HD 6912
#define E  8
#define TK 2
#define CAP 384            // ceil(1.5*2048/8)

typedef __attribute__((ext_vector_type(16))) __bf16 v16bf;
typedef __attribute__((ext_vector_type(8)))  float  v8f;
typedef unsigned short ushort_t;
typedef int v4i __attribute__((vector_size(16)));   // matches builtin's int4 param

#define AS1 __attribute__((address_space(1)))
#define AS3 __attribute__((address_space(3)))

#if defined(__has_builtin)
#if __has_builtin(__builtin_amdgcn_global_load_async_to_lds_b128) && \
    __has_builtin(__builtin_amdgcn_s_wait_asynccnt)
#define HAVE_ASYNC 1
#endif
#endif

#if defined(HAVE_ASYNC)
// async DMA: 16 bytes per lane, global -> LDS, tracked by ASYNCcnt
#define ASYNC_CP16(ldsp, gp) \
  __builtin_amdgcn_global_load_async_to_lds_b128((AS1 v4i*)(gp), (AS3 v4i*)(ldsp), 0, 0)
#endif

__device__ __forceinline__ float bf2f(ushort_t u) {
  union { unsigned int i; float f; } v; v.i = ((unsigned int)u) << 16; return v.f;
}
__device__ __forceinline__ ushort_t f2bf(float f) {
  union { float f; unsigned int i; } v; v.f = f;
  unsigned int r = v.i + 0x7FFFu + ((v.i >> 16) & 1u);   // RNE
  return (ushort_t)(r >> 16);
}

// ---------------- 1. router: logits, top-2, masked softmax, z ----------------
__global__ __launch_bounds__(256) void router_kernel(
    const ushort_t* __restrict__ inp,      // [G,S,M] bf16
    const ushort_t* __restrict__ gate,     // [M,E]   bf16
    int*   __restrict__ eidx,              // [G*S,2]
    float* __restrict__ prob,              // [G*S,2] (bf16-rounded values)
    float* __restrict__ zbuf)              // [G*S]
{
  const int lane = threadIdx.x & 31;
  const int wid  = threadIdx.x >> 5;
  const int token = blockIdx.x * 8 + wid;
  if (token >= G * S) return;
  const ushort_t* x = inp + (size_t)token * MD;

  float acc[E] = {};
  for (int m = lane; m < MD; m += 32) {
    float xv = bf2f(x[m]);
    const uint4 q = *(const uint4*)(gate + (size_t)m * E);   // 8 bf16
    const ushort_t* gv = (const ushort_t*)&q;
#pragma unroll
    for (int e = 0; e < E; ++e) acc[e] += xv * bf2f(gv[e]);
  }
#pragma unroll
  for (int off = 16; off > 0; off >>= 1)
#pragma unroll
    for (int e = 0; e < E; ++e) acc[e] += __shfl_down(acc[e], off, 32);

  if (lane == 0) {
    int i0 = 0;
#pragma unroll
    for (int e = 1; e < E; ++e) if (acc[e] > acc[i0]) i0 = e;
    int i1 = (i0 == 0) ? 1 : 0;
#pragma unroll
    for (int e = 0; e < E; ++e) if (e != i0 && acc[e] > acc[i1]) i1 = e;
    float mx = acc[i0];
    float e1 = __expf(acc[i1] - mx);
    float den = 1.0f + e1;                 // other experts masked to -3.4e38 -> exp==0
    float zs = 0.0f;
#pragma unroll
    for (int e = 0; e < E; ++e) zs += __expf(acc[e] - mx);
    eidx[token * 2 + 0] = i0;
    eidx[token * 2 + 1] = i1;
    prob[token * 2 + 0] = bf2f(f2bf(1.0f / den));   // reference casts probs to bf16
    prob[token * 2 + 1] = bf2f(f2bf(e1 / den));
    zbuf[token] = __logf(zs) + mx;
  }
}

// ---------------- 2. ordered dispatch scan (rank-0 before rank-1) ----------------
__global__ void priority_kernel(const int* __restrict__ eidx,
                                int* __restrict__ slot,     // [G*S,2] slot or -1
                                int* __restrict__ rev)      // [G,E,CAP] src token or -1
{
  if (threadIdx.x != 0) return;
  const int g = blockIdx.x;
  int* revg = rev + (size_t)g * E * CAP;
  for (int i = 0; i < E * CAP; ++i) revg[i] = -1;
  int cnt[E] = {};
  for (int k = 0; k < TK; ++k)
    for (int s = 0; s < S; ++s) {
      int t = (g * S + s) * 2 + k;
      int e = eidx[t];
      int c = cnt[e]++;
      if (c < CAP) { slot[t] = c; revg[e * CAP + c] = s; }
      else         { slot[t] = -1; }
    }
}

// ---------------- 3a. per-(g,e) load-balance sums (deterministic tree) ----------------
__global__ __launch_bounds__(256) void loss_ge_kernel(
    const int* __restrict__ eidx, const float* __restrict__ prob,
    float* __restrict__ tok_ge, float* __restrict__ prob_ge)
{
  const int ge = blockIdx.x, g = ge / E, e = ge % E;
  __shared__ float sc[256], sp[256];
  float c = 0.f, p = 0.f;
  for (int s = threadIdx.x; s < S; s += 256) {
    int b = (g * S + s) * 2;
    if (eidx[b + 0] == e) { c += 1.f; p += prob[b + 0]; }
    if (eidx[b + 1] == e) { c += 1.f; p += prob[b + 1]; }
  }
  sc[threadIdx.x] = c; sp[threadIdx.x] = p; __syncthreads();
  for (int o = 128; o > 0; o >>= 1) {
    if (threadIdx.x < o) { sc[threadIdx.x] += sc[threadIdx.x + o]; sp[threadIdx.x] += sp[threadIdx.x + o]; }
    __syncthreads();
  }
  if (threadIdx.x == 0) { tok_ge[ge] = sc[0] / S; prob_ge[ge] = sp[0] / S; }
}

// ---------------- 3b. final scalar loss ----------------
__global__ __launch_bounds__(256) void loss_final_kernel(
    const float* __restrict__ tok_ge, const float* __restrict__ prob_ge,
    const float* __restrict__ zbuf, void* aux_out, int mode)
{
  __shared__ float sz[256];
  float z2 = 0.f;
  for (int i = threadIdx.x; i < G * S; i += 256) { float z = zbuf[i]; z2 += z * z; }
  sz[threadIdx.x] = z2; __syncthreads();
  for (int o = 128; o > 0; o >>= 1) {
    if (threadIdx.x < o) sz[threadIdx.x] += sz[threadIdx.x + o];
    __syncthreads();
  }
  if (threadIdx.x == 0) {
    float ap = 0.f;
    for (int i = 0; i < G * E; ++i) ap += tok_ge[i] * prob_ge[i];
    ap /= (float)(G * E);
    float aux = 0.01f * ap * (float)(E * E) + 0.01f * (sz[0] / (float)(G * S));
    if (mode == 0)      *(ushort_t*)aux_out = f2bf(aux);
    else if (mode == 1) *(float*)aux_out    = aux;
  }
}

// ---------------- 4. gather expert inputs [G*E, CAP, MD] ----------------
__global__ __launch_bounds__(256) void gather_kernel(
    const ushort_t* __restrict__ inp, const int* __restrict__ rev,
    ushort_t* __restrict__ xg)
{
  const size_t idx = (size_t)blockIdx.x * blockDim.x + threadIdx.x;  // 8 halves each
  const size_t row = idx / (MD / 8);
  const int chunk  = (int)(idx % (MD / 8));
  const int g = (int)(row / (E * CAP));
  const int s = rev[row];
  uint4 v;
  if (s >= 0) v = *(const uint4*)(inp + ((size_t)g * S + s) * MD + (size_t)chunk * 8);
  else        v = make_uint4(0u, 0u, 0u, 0u);
  *(uint4*)(xg + row * MD + (size_t)chunk * 8) = v;
}

// ---------------- WMMA GEMM tiles ----------------
#define BM 128
#define BN 64
#define BK 32

union Frag { v16bf v; uint4 q[2]; };

// 5. fused SwiGLU:  hid = silu(X*Wg) .* (X*W0)   per (g,e)
__global__ __launch_bounds__(256) void gemm1_swiglu_kernel(
    const ushort_t* __restrict__ xg,   // [G*E, CAP, MD]
    const ushort_t* __restrict__ wgm,  // [E, MD, HD]
    const ushort_t* __restrict__ w0m,  // [E, MD, HD]
    ushort_t* __restrict__ hid)        // [G*E, CAP, HD]
{
  const int ge = blockIdx.z, e = ge % E;
  const int bm0 = blockIdx.y * BM, bn0 = blockIdx.x * BN;
  const ushort_t* X  = xg  + (size_t)ge * CAP * MD;
  const ushort_t* Bg = wgm + (size_t)e * MD * HD;
  const ushort_t* B0 = w0m + (size_t)e * MD * HD;

  const int t = threadIdx.x, lane = t & 31, wid = t >> 5;
  const int wm = wid & 3, wn = wid >> 2;          // 4x2 waves, 32x32 per wave
  const int ar = t >> 1,  ac = (t & 1) * 16;      // A staging: 32B/thread
  const int br = t >> 3,  bc = (t & 7) * 8;       // B staging: 16B/thread
  const int arow = lane & 15, kb = (lane < 16) ? 0 : 8;

  v8f accG[2][2] = {}, acc0[2][2] = {};

#if defined(HAVE_ASYNC)
  // ---- async double-buffered pipeline: DMA stage i+2 overlaps compute of i+1 ----
  __shared__ ushort_t sA[2][BM][BK];
  __shared__ ushort_t sBg[2][BK][BN];
  __shared__ ushort_t sB0[2][BK][BN];
  const int NK = MD / BK;

#define G1_ISSUE(k0, buf)                                                        \
  do {                                                                           \
    ASYNC_CP16(&sA[buf][ar][ac],     X  + (size_t)(bm0 + ar) * MD + (k0) + ac);  \
    ASYNC_CP16(&sA[buf][ar][ac + 8], X  + (size_t)(bm0 + ar) * MD + (k0) + ac + 8); \
    ASYNC_CP16(&sBg[buf][br][bc],    Bg + (size_t)((k0) + br) * HD + bn0 + bc);  \
    ASYNC_CP16(&sB0[buf][br][bc],    B0 + (size_t)((k0) + br) * HD + bn0 + bc);  \
  } while (0)

  G1_ISSUE(0, 0);
  G1_ISSUE(BK, 1);

  for (int i = 0; i < NK; ++i) {
    const int cur = i & 1;
    if (i + 1 < NK) __builtin_amdgcn_s_wait_asynccnt(4);  // stage i landed, i+1 in flight
    else            __builtin_amdgcn_s_wait_asynccnt(0);
    __syncthreads();

    Frag a[2], bg[2], b0[2];
#pragma unroll
    for (int ti = 0; ti < 2; ++ti) {
      int r = wm * 32 + ti * 16 + arow;
      a[ti].q[0] = *(const uint4*)&sA[cur][r][kb];
      a[ti].q[1] = *(const uint4*)&sA[cur][r][kb + 16];
    }
#pragma unroll
    for (int tj = 0; tj < 2; ++tj) {
      int cn = wn * 32 + tj * 16;
      for (int s = 0; s < 16; ++s) {
        int kk = kb + (s < 8 ? s : s + 8);
        ((ushort_t*)&bg[tj])[s] = sBg[cur][kk][cn + arow];
        ((ushort_t*)&b0[tj])[s] = sB0[cur][kk][cn + arow];
      }
    }
#pragma unroll
    for (int ti = 0; ti < 2; ++ti)
#pragma unroll
      for (int tj = 0; tj < 2; ++tj) {
        accG[ti][tj] = __builtin_amdgcn_wmma_f32_16x16x32_bf16(
            false, a[ti].v, false, bg[tj].v, (short)0, accG[ti][tj], false, false);
        acc0[ti][tj] = __builtin_amdgcn_wmma_f32_16x16x32_bf16(
            false, a[ti].v, false, b0[tj].v, (short)0, acc0[ti][tj], false, false);
      }
    __syncthreads();                       // everyone done reading buffer `cur`
    if (i + 2 < NK) G1_ISSUE((i + 2) * BK, cur);
  }
#undef G1_ISSUE
#else
  // ---- fallback: synchronous single-buffer staging ----
  __shared__ ushort_t sA[BM][BK];
  __shared__ ushort_t sBg[BK][BN];
  __shared__ ushort_t sB0[BK][BN];

  for (int k0 = 0; k0 < MD; k0 += BK) {
    const uint4* gA = (const uint4*)(X + (size_t)(bm0 + ar) * MD + k0 + ac);
    *(uint4*)&sA[ar][ac]     = gA[0];
    *(uint4*)&sA[ar][ac + 8] = gA[1];
    *(uint4*)&sBg[br][bc] = *(const uint4*)(Bg + (size_t)(k0 + br) * HD + bn0 + bc);
    *(uint4*)&sB0[br][bc] = *(const uint4*)(B0 + (size_t)(k0 + br) * HD + bn0 + bc);
    if (k0 + BK < MD)
      __builtin_prefetch(Bg + (size_t)(k0 + BK + br) * HD + bn0 + bc, 0, 1);
    __syncthreads();

    Frag a[2], bg[2], b0[2];
#pragma unroll
    for (int ti = 0; ti < 2; ++ti) {
      int r = wm * 32 + ti * 16 + arow;
      a[ti].q[0] = *(const uint4*)&sA[r][kb];
      a[ti].q[1] = *(const uint4*)&sA[r][kb + 16];
    }
#pragma unroll
    for (int tj = 0; tj < 2; ++tj) {
      int cn = wn * 32 + tj * 16;
      for (int s = 0; s < 16; ++s) {
        int kk = kb + (s < 8 ? s : s + 8);
        ((ushort_t*)&bg[tj])[s] = sBg[kk][cn + arow];
        ((ushort_t*)&b0[tj])[s] = sB0[kk][cn + arow];
      }
    }
#pragma unroll
    for (int ti = 0; ti < 2; ++ti)
#pragma unroll
      for (int tj = 0; tj < 2; ++tj) {
        accG[ti][tj] = __builtin_amdgcn_wmma_f32_16x16x32_bf16(
            false, a[ti].v, false, bg[tj].v, (short)0, accG[ti][tj], false, false);
        acc0[ti][tj] = __builtin_amdgcn_wmma_f32_16x16x32_bf16(
            false, a[ti].v, false, b0[tj].v, (short)0, acc0[ti][tj], false, false);
      }
    __syncthreads();
  }
#endif

  ushort_t* Hout = hid + (size_t)ge * CAP * HD;
  const int ncol = lane & 15, mofs = (lane < 16) ? 0 : 8;
#pragma unroll
  for (int ti = 0; ti < 2; ++ti)
#pragma unroll
    for (int tj = 0; tj < 2; ++tj) {
      int n = bn0 + wn * 32 + tj * 16 + ncol;
#pragma unroll
      for (int r = 0; r < 8; ++r) {
        int m = bm0 + wm * 32 + ti * 16 + mofs + r;
        float gv = accG[ti][tj][r];
        float ov = acc0[ti][tj][r];
        float sg = gv / (1.0f + __expf(-gv));     // silu
        Hout[(size_t)m * HD + n] = f2bf(sg * ov);
      }
    }
}

// 6. expert_out = hidden * Wo     per (g,e)
__global__ __launch_bounds__(256) void gemm2_kernel(
    const ushort_t* __restrict__ hid,  // [G*E, CAP, HD]
    const ushort_t* __restrict__ wom,  // [E, HD, MD]
    ushort_t* __restrict__ eout)       // [G*E, CAP, MD]
{
  const int ge = blockIdx.z, e = ge % E;
  const int bm0 = blockIdx.y * BM, bn0 = blockIdx.x * BN;
  const ushort_t* A = hid + (size_t)ge * CAP * HD;
  const ushort_t* B = wom + (size_t)e * HD * MD;

  const int t = threadIdx.x, lane = t & 31, wid = t >> 5;
  const int wm = wid & 3, wn = wid >> 2;
  const int ar = t >> 1,  ac = (t & 1) * 16;
  const int br = t >> 3,  bc = (t & 7) * 8;
  const int arow = lane & 15, kb = (lane < 16) ? 0 : 8;

  v8f acc[2][2] = {};

#if defined(HAVE_ASYNC)
  __shared__ ushort_t sA[2][BM][BK];
  __shared__ ushort_t sB[2][BK][BN];
  const int NK = HD / BK;

#define G2_ISSUE(k0, buf)                                                       \
  do {                                                                          \
    ASYNC_CP16(&sA[buf][ar][ac],     A + (size_t)(bm0 + ar) * HD + (k0) + ac);  \
    ASYNC_CP16(&sA[buf][ar][ac + 8], A + (size_t)(bm0 + ar) * HD + (k0) + ac + 8); \
    ASYNC_CP16(&sB[buf][br][bc],     B + (size_t)((k0) + br) * MD + bn0 + bc);  \
  } while (0)

  G2_ISSUE(0, 0);
  G2_ISSUE(BK, 1);

  for (int i = 0; i < NK; ++i) {
    const int cur = i & 1;
    if (i + 1 < NK) __builtin_amdgcn_s_wait_asynccnt(3);
    else            __builtin_amdgcn_s_wait_asynccnt(0);
    __syncthreads();

    Frag a[2], b[2];
#pragma unroll
    for (int ti = 0; ti < 2; ++ti) {
      int r = wm * 32 + ti * 16 + arow;
      a[ti].q[0] = *(const uint4*)&sA[cur][r][kb];
      a[ti].q[1] = *(const uint4*)&sA[cur][r][kb + 16];
    }
#pragma unroll
    for (int tj = 0; tj < 2; ++tj) {
      int cn = wn * 32 + tj * 16;
      for (int s = 0; s < 16; ++s)
        ((ushort_t*)&b[tj])[s] = sB[cur][kb + (s < 8 ? s : s + 8)][cn + arow];
    }
#pragma unroll
    for (int ti = 0; ti < 2; ++ti)
#pragma unroll
      for (int tj = 0; tj < 2; ++tj)
        acc[ti][tj] = __builtin_amdgcn_wmma_f32_16x16x32_bf16(
            false, a[ti].v, false, b[tj].v, (short)0, acc[ti][tj], false, false);
    __syncthreads();
    if (i + 2 < NK) G2_ISSUE((i + 2) * BK, cur);
  }
#undef G2_ISSUE
#else
  __shared__ ushort_t sA[BM][BK];
  __shared__ ushort_t sB[BK][BN];

  for (int k0 = 0; k0 < HD; k0 += BK) {
    const uint4* gA = (const uint4*)(A + (size_t)(bm0 + ar) * HD + k0 + ac);
    *(uint4*)&sA[ar][ac]     = gA[0];
    *(uint4*)&sA[ar][ac + 8] = gA[1];
    *(uint4*)&sB[br][bc] = *(const uint4*)(B + (size_t)(k0 + br) * MD + bn0 + bc);
    if (k0 + BK < HD)
      __builtin_prefetch(B + (size_t)(k0 + BK + br) * MD + bn0 + bc, 0, 1);
    __syncthreads();

    Frag a[2], b[2];
#pragma unroll
    for (int ti = 0; ti < 2; ++ti) {
      int r = wm * 32 + ti * 16 + arow;
      a[ti].q[0] = *(const uint4*)&sA[r][kb];
      a[ti].q[1] = *(const uint4*)&sA[r][kb + 16];
    }
#pragma unroll
    for (int tj = 0; tj < 2; ++tj) {
      int cn = wn * 32 + tj * 16;
      for (int s = 0; s < 16; ++s)
        ((ushort_t*)&b[tj])[s] = sB[kb + (s < 8 ? s : s + 8)][cn + arow];
    }
#pragma unroll
    for (int ti = 0; ti < 2; ++ti)
#pragma unroll
      for (int tj = 0; tj < 2; ++tj)
        acc[ti][tj] = __builtin_amdgcn_wmma_f32_16x16x32_bf16(
            false, a[ti].v, false, b[tj].v, (short)0, acc[ti][tj], false, false);
    __syncthreads();
  }
#endif

  ushort_t* O = eout + (size_t)ge * CAP * MD;
  const int ncol = lane & 15, mofs = (lane < 16) ? 0 : 8;
#pragma unroll
  for (int ti = 0; ti < 2; ++ti)
#pragma unroll
    for (int tj = 0; tj < 2; ++tj) {
      int n = bn0 + wn * 32 + tj * 16 + ncol;
#pragma unroll
      for (int r = 0; r < 8; ++r) {
        int m = bm0 + wm * 32 + ti * 16 + mofs + r;
        O[(size_t)m * MD + n] = f2bf(acc[ti][tj][r]);
      }
    }
}

// ---------------- 7. combine: out[g,s,:] = sum_k p_k * eout[g,e_k,c_k,:] ----------------
__global__ __launch_bounds__(256) void combine_kernel(
    const ushort_t* __restrict__ eout, const int* __restrict__ slot,
    const int* __restrict__ eidx, const float* __restrict__ prob,
    ushort_t* __restrict__ out)
{
  const size_t idx = (size_t)blockIdx.x * blockDim.x + threadIdx.x;  // 8 halves each
  const size_t token = idx / (MD / 8);
  const int chunk = (int)(idx % (MD / 8));
  const int g = (int)(token / S);
  float acc[8] = {};
#pragma unroll
  for (int k = 0; k < TK; ++k) {
    int b = (int)token * 2 + k;
    int c = slot[b];
    if (c < 0) continue;
    int e = eidx[b];
    float p = prob[b];
    const uint4 v = *(const uint4*)(eout + (((size_t)(g * E + e)) * CAP + c) * MD + (size_t)chunk * 8);
    const ushort_t* hv = (const ushort_t*)&v;
#pragma unroll
    for (int i = 0; i < 8; ++i) acc[i] += p * bf2f(hv[i]);
  }
  ushort_t o[8];
#pragma unroll
  for (int i = 0; i < 8; ++i) o[i] = f2bf(acc[i]);
  *(uint4*)(out + token * MD + (size_t)chunk * 8) = *(const uint4*)o;
}

// ---------------- host ----------------
extern "C" void kernel_launch(void* const* d_in, const int* in_sizes, int n_in,
                              void* d_out, int out_size, void* d_ws, size_t ws_size,
                              hipStream_t stream) {
  const ushort_t* inp  = (const ushort_t*)d_in[0];   // [G,S,MD]
  const ushort_t* gate = (const ushort_t*)d_in[1];   // [MD,E]
  const ushort_t* wg   = (const ushort_t*)d_in[2];   // [E,MD,HD]
  const ushort_t* w0   = (const ushort_t*)d_in[3];   // [E,MD,HD]
  const ushort_t* wo   = (const ushort_t*)d_in[4];   // [E,HD,MD]
  ushort_t* out = (ushort_t*)d_out;

  char* ws = (char*)d_ws;
  size_t off = 0;
  auto alloc = [&](size_t bytes) { char* p = ws + off; off += (bytes + 255) & ~(size_t)255; return p; };
  ushort_t* xg   = (ushort_t*)alloc((size_t)G * E * CAP * MD * 2);
  ushort_t* hid  = (ushort_t*)alloc((size_t)G * E * CAP * HD * 2);
  ushort_t* eout = (ushort_t*)alloc((size_t)G * E * CAP * MD * 2);
  int*   eidx    = (int*)  alloc((size_t)G * S * TK * 4);
  int*   slot    = (int*)  alloc((size_t)G * S * TK * 4);
  float* prob    = (float*)alloc((size_t)G * S * TK * 4);
  float* zbuf    = (float*)alloc((size_t)G * S * 4);
  int*   rev     = (int*)  alloc((size_t)G * E * CAP * 4);
  float* tok_ge  = (float*)alloc((size_t)G * E * 4);
  float* prob_ge = (float*)alloc((size_t)G * E * 4);
  (void)ws_size; (void)in_sizes; (void)n_in;

  // 1. router: one wave per token
  router_kernel<<<(G * S) / 8, 256, 0, stream>>>(inp, gate, eidx, prob, zbuf);
  // 2. ordered capacity scan
  priority_kernel<<<G, 32, 0, stream>>>(eidx, slot, rev);
  // 3a. load-balance sums
  loss_ge_kernel<<<G * E, 256, 0, stream>>>(eidx, prob, tok_ge, prob_ge);
  // 4. gather
  {
    size_t total = (size_t)G * E * CAP * (MD / 8);
    gather_kernel<<<(unsigned)(total / 256), 256, 0, stream>>>(inp, rev, xg);
  }
  // 5. fused SwiGLU GEMM:  grid (HD/BN, CAP/BM, G*E)
  gemm1_swiglu_kernel<<<dim3(HD / BN, CAP / BM, G * E), 256, 0, stream>>>(xg, wg, w0, hid);
  // 6. output GEMM: grid (MD/BN, CAP/BM, G*E)
  gemm2_kernel<<<dim3(MD / BN, CAP / BM, G * E), 256, 0, stream>>>(hid, wo, eout);
  // 7. combine
  {
    size_t total = (size_t)G * S * (MD / 8);
    combine_kernel<<<(unsigned)(total / 256), 256, 0, stream>>>(eout, slot, eidx, prob, out);
  }
  // 3b. final loss -> tail of d_out
  {
    const size_t GSM = (size_t)G * S * MD;
    int mode = -1;
    void* auxp = (void*)(out + GSM);
    if ((size_t)out_size == GSM + 1) mode = 0;        // bf16 scalar appended
    else if ((size_t)out_size >= GSM + 2) mode = 1;   // room for f32 scalar
    loss_final_kernel<<<1, 256, 0, stream>>>(tok_ge, prob_ge, zbuf, auxp, mode);
  }
}
